// SIGFormerBlock_69818988364302
// MI455X (gfx1250) — matrix operand order repaired and checked by reading, not compiled, not run on hardware
//
#include <hip/hip_runtime.h>
#include <hip/hip_bf16.h>

typedef __attribute__((ext_vector_type(16))) _Float16 v16h;
typedef __attribute__((ext_vector_type(8)))  float    v8f;

#define B_  2
#define N_  512
#define D_  128
#define H_  4
#define HT_ 4
#define FF_ 128

// ---------------------------------------------------------------------------
// WMMA GEMM, 32x32 C-tile per wave (4 x v_wmma_f32_16x16x32_f16 per K-step).
// C[M,N] = relu?( alpha * A@B(^T) + bias[n] + beta*Res )
// K multiple of 32, M and N multiples of 32. A/B K order is contiguous: k = 16*half + i.
// ---------------------------------------------------------------------------
template <int TRANSB>
__global__ __launch_bounds__(32) void gemm_wmma_kernel(
    const float* __restrict__ A, int lda,
    const float* __restrict__ B, int ldb,
    float* __restrict__ C, int ldc,
    int K, float alpha,
    const float* __restrict__ bias,
    const float* __restrict__ res, int ldr, float beta,
    int relu)
{
    const int lane = threadIdx.x & 31;
    const int lo = lane & 15;     // M index (A,C) / N index (B,C)
    const int hi = lane >> 4;     // lane-half selector
    const int m0 = blockIdx.y * 32;
    const int n0 = blockIdx.x * 32;

    v8f acc00 = {}; v8f acc01 = {}; v8f acc10 = {}; v8f acc11 = {};

    const float* Arow0 = A + (size_t)(m0 + lo) * lda;
    const float* Arow1 = Arow0 + (size_t)16 * lda;

    for (int k0 = 0; k0 < K; k0 += 32) {
        // speculative prefetch of next A tile (dropped silently if OOB)
        __builtin_prefetch(Arow0 + k0 + 32, 0, 1);

        // A fragments: 16x32 f16, contiguous K-pairs -> float2 loads.
        // VGPR v covers K = 2v + 16*half + {0,1}
        v16h a0, a1;
#pragma unroll
        for (int v = 0; v < 8; ++v) {
            int base = 2 * v + 16 * hi;
            float2 p0 = *(const float2*)(Arow0 + k0 + base);
            float2 p1 = *(const float2*)(Arow1 + k0 + base);
            a0[2 * v]     = (_Float16)p0.x;
            a0[2 * v + 1] = (_Float16)p0.y;
            a1[2 * v]     = (_Float16)p1.x;
            a1[2 * v + 1] = (_Float16)p1.y;
        }

        // B fragments: 32x16 f16, element i covers K = k0 + i + 16*half
        v16h b0, b1;
        if constexpr (TRANSB) {
            // B is [N,K] row-major: 16 contiguous K floats -> float4 loads
            const float* Br0 = B + (size_t)(n0 + lo) * ldb + k0 + 16 * hi;
            const float* Br1 = Br0 + (size_t)16 * ldb;
#pragma unroll
            for (int q = 0; q < 4; ++q) {
                float4 r0 = *(const float4*)(Br0 + 4 * q);
                float4 r1 = *(const float4*)(Br1 + 4 * q);
                b0[4 * q + 0] = (_Float16)r0.x; b0[4 * q + 1] = (_Float16)r0.y;
                b0[4 * q + 2] = (_Float16)r0.z; b0[4 * q + 3] = (_Float16)r0.w;
                b1[4 * q + 0] = (_Float16)r1.x; b1[4 * q + 1] = (_Float16)r1.y;
                b1[4 * q + 2] = (_Float16)r1.z; b1[4 * q + 3] = (_Float16)r1.w;
            }
        } else {
            // B is [K,N] row-major: strided column reads
            const float* Bc0 = B + (size_t)(k0 + 16 * hi) * ldb + n0 + lo;
#pragma unroll
            for (int i = 0; i < 16; ++i) {
                b0[i] = (_Float16)Bc0[(size_t)i * ldb];
                b1[i] = (_Float16)Bc0[(size_t)i * ldb + 16];
            }
        }

        acc00 = __builtin_amdgcn_wmma_f32_16x16x32_f16(false, a0, false, b0, (short)0, acc00, false, false);
        acc01 = __builtin_amdgcn_wmma_f32_16x16x32_f16(false, a0, false, b1, (short)0, acc01, false, false);
        acc10 = __builtin_amdgcn_wmma_f32_16x16x32_f16(false, a1, false, b0, (short)0, acc10, false, false);
        acc11 = __builtin_amdgcn_wmma_f32_16x16x32_f16(false, a1, false, b1, (short)0, acc11, false, false);
    }

    // Epilogue. Bias is column-invariant across the 8 rows of an accumulator:
    // load it once; hoist the residual check to one uniform branch per tile.
    float bv0 = 0.f, bv1 = 0.f;
    if (bias) { bv0 = bias[n0 + lo]; bv1 = bias[n0 + 16 + lo]; }

    auto emit = [&](const v8f& acc, int mbase, int nn, float badd) {
        const int n = nn + lo;
        if (res) {
#pragma unroll
            for (int r = 0; r < 8; ++r) {
                int m = mbase + r + 8 * hi;
                float v = fmaf(alpha, acc[r], badd) + beta * res[(size_t)m * ldr + n];
                v = relu ? fmaxf(v, 0.f) : v;
                C[(size_t)m * ldc + n] = v;
            }
        } else {
#pragma unroll
            for (int r = 0; r < 8; ++r) {
                int m = mbase + r + 8 * hi;
                float v = fmaf(alpha, acc[r], badd);
                v = relu ? fmaxf(v, 0.f) : v;
                C[(size_t)m * ldc + n] = v;
            }
        }
    };
    emit(acc00, m0,      n0,      bv0);
    emit(acc01, m0,      n0 + 16, bv1);
    emit(acc10, m0 + 16, n0,      bv0);
    emit(acc11, m0 + 16, n0 + 16, bv1);
}

// ---------------------------------------------------------------------------
// Residual + LayerNorm over D=128: out = LN(x + r) * s + b
// ---------------------------------------------------------------------------
__global__ void ln_residual_kernel(const float* __restrict__ x,
                                   const float* __restrict__ r,
                                   const float* __restrict__ s,
                                   const float* __restrict__ b,
                                   float* __restrict__ out)
{
    __shared__ float red[128];
    int row = blockIdx.x, t = threadIdx.x;
    float v = x[(size_t)row * 128 + t];
    if (r) v += r[(size_t)row * 128 + t];
    red[t] = v; __syncthreads();
    for (int s2 = 64; s2 > 0; s2 >>= 1) { if (t < s2) red[t] += red[t + s2]; __syncthreads(); }
    float mean = red[0] * (1.f / 128.f); __syncthreads();
    float d = v - mean;
    red[t] = d * d; __syncthreads();
    for (int s2 = 64; s2 > 0; s2 >>= 1) { if (t < s2) red[t] += red[t + s2]; __syncthreads(); }
    float var = red[0] * (1.f / 128.f);
    out[(size_t)row * 128 + t] = d * rsqrtf(var + 1e-5f) * s[t] + b[t];
}

// ---------------------------------------------------------------------------
// Row softmax over 512 cols, optional per-row pre-scale (mod may be negative)
// ---------------------------------------------------------------------------
__global__ void softmax_row_kernel(float* __restrict__ S, const float* __restrict__ mod)
{
    __shared__ float red[256];
    int row = blockIdx.x, t = threadIdx.x;
    float* p = S + (size_t)row * 512;
    float m = mod ? mod[row] : 1.f;
    float a = p[t] * m;
    float b = p[t + 256] * m;
    red[t] = fmaxf(a, b); __syncthreads();
    for (int s = 128; s > 0; s >>= 1) { if (t < s) red[t] = fmaxf(red[t], red[t + s]); __syncthreads(); }
    float mx = red[0]; __syncthreads();
    a = __expf(a - mx); b = __expf(b - mx);
    red[t] = a + b; __syncthreads();
    for (int s = 128; s > 0; s >>= 1) { if (t < s) red[t] += red[t + s]; __syncthreads(); }
    float inv = 1.f / red[0];
    p[t] = a * inv; p[t + 256] = b * inv;
}

// mean over 512 columns
__global__ void row_mean512_kernel(const float* __restrict__ A, float* __restrict__ out)
{
    __shared__ float red[256];
    int row = blockIdx.x, t = threadIdx.x;
    const float* p = A + (size_t)row * 512;
    red[t] = p[t] + p[t + 256]; __syncthreads();
    for (int s = 128; s > 0; s >>= 1) { if (t < s) red[t] += red[t + s]; __syncthreads(); }
    if (t == 0) out[row] = red[0] * (1.f / 512.f);
}

// mod[b,h,n] = mean_dd( A_mean[b,n]*adjA_w[h*32+dd] + adjA_b[h*32+dd] )
__global__ void mod_kernel(const float* __restrict__ Amean,
                           const float* __restrict__ aw, const float* __restrict__ ab,
                           float* __restrict__ mod)
{
    int idx = blockIdx.x * blockDim.x + threadIdx.x;    // B*H*N = 4096
    if (idx >= B_ * H_ * N_) return;
    int n = idx & 511, h = (idx >> 9) & 3, b = idx >> 11;
    float am = Amean[b * N_ + n];
    float acc = 0.f;
#pragma unroll
    for (int dd = 0; dd < 32; ++dd) { int d = h * 32 + dd; acc += am * aw[d] + ab[d]; }
    mod[idx] = acc * (1.f / 32.f);
}

// s[row] = sum_d X[row,d] * w[d]   (Linear(D,1), no bias)
__global__ void rowdot_kernel(const float* __restrict__ X, const float* __restrict__ w,
                              float* __restrict__ out)
{
    __shared__ float red[128];
    int row = blockIdx.x, t = threadIdx.x;
    red[t] = X[(size_t)row * 128 + t] * w[t]; __syncthreads();
    for (int s = 64; s > 0; s >>= 1) { if (t < s) red[t] += red[t + s]; __syncthreads(); }
    if (t == 0) out[row] = red[0];
}

// adj_out[b,i,j] = sum_f relu((A_q*s[b,j]+ob) * w1[f] + b1[f]) * w2[f] + b2
__global__ void adj_out_kernel(const float* __restrict__ A_q, const float* __restrict__ s,
                               const float* __restrict__ obp,
                               const float* __restrict__ w1, const float* __restrict__ b1,
                               const float* __restrict__ w2, const float* __restrict__ b2p,
                               float* __restrict__ out)
{
    int idx = blockIdx.x * blockDim.x + threadIdx.x;    // B*N*N = 524288
    if (idx >= B_ * N_ * N_) return;
    int j = idx & 511;
    int b = idx >> 18;
    float a = A_q[idx] * s[b * N_ + j] + obp[0];
    float acc = b2p[0];
#pragma unroll 8
    for (int f = 0; f < FF_; ++f) {
        float t = a * w1[f] + b1[f];
        acc += (t > 0.f) ? t * w2[f] : 0.f;
    }
    out[idx] = acc;
}

// feat[b,n, d*5+m] from the 5 diffusion matrices (stack axis=-1 then reshape)
__global__ void interleave_feat_kernel(const float* __restrict__ X,
                                       const float* __restrict__ aq1, const float* __restrict__ aq2,
                                       const float* __restrict__ ah1, const float* __restrict__ ah2,
                                       float* __restrict__ feat)
{
    int idx = blockIdx.x * blockDim.x + threadIdx.x;    // B*N*D = 131072
    if (idx >= B_ * N_ * D_) return;
    int d = idx & 127;
    int bn = idx >> 7;
    size_t base = (size_t)bn * (D_ * 5) + (size_t)d * 5;
    feat[base + 0] = X[idx];
    feat[base + 1] = aq1[idx];
    feat[base + 2] = aq2[idx];
    feat[base + 3] = ah1[idx];
    feat[base + 4] = ah2[idx];
}

// [H,F,O] -> [F, H*O]
__global__ void pack_hfo_kernel(const float* __restrict__ in, float* __restrict__ out,
                                int H, int F, int O)
{
    int idx = blockIdx.x * blockDim.x + threadIdx.x;
    int total = H * F * O;
    if (idx >= total) return;
    int o = idx % O;
    int f = (idx / O) % F;
    int h = idx / (O * F);
    out[(size_t)f * (H * O) + h * O + o] = in[idx];
}

// ---------------------------------------------------------------------------
static void launch_gemm(hipStream_t st, const float* A, int lda,
                        const float* B, int ldb, int transB,
                        float* C, int ldc, int M, int N, int K, float alpha,
                        const float* bias, const float* res, int ldr, float beta, int relu)
{
    dim3 g(N / 32, M / 32, 1);
    if (transB)
        gemm_wmma_kernel<1><<<g, 32, 0, st>>>(A, lda, B, ldb, C, ldc, K, alpha, bias, res, ldr, beta, relu);
    else
        gemm_wmma_kernel<0><<<g, 32, 0, st>>>(A, lda, B, ldb, C, ldc, K, alpha, bias, res, ldr, beta, relu);
}

extern "C" void kernel_launch(void* const* d_in, const int* in_sizes, int n_in,
                              void* d_out, int out_size, void* d_ws, size_t ws_size,
                              hipStream_t stream)
{
    (void)in_sizes; (void)n_in; (void)out_size; (void)ws_size;

    const float* X    = (const float*)d_in[0];
    const float* A_q  = (const float*)d_in[1];
    const float* A_h  = (const float*)d_in[2];
    const float* dgcn_theta = (const float*)d_in[3];
    const float* dgcn_bias  = (const float*)d_in[4];   // [4,128] == flat [512] head-major
    const float* dgcn_ow    = (const float*)d_in[5];
    const float* dgcn_ob    = (const float*)d_in[6];
    const float* wq = (const float*)d_in[7];  const float* bq = (const float*)d_in[8];
    const float* wk = (const float*)d_in[9];  const float* bk = (const float*)d_in[10];
    const float* wv = (const float*)d_in[11]; const float* bv = (const float*)d_in[12];
    const float* adjA_w = (const float*)d_in[13]; const float* adjA_b = (const float*)d_in[14];
    const float* node_ow = (const float*)d_in[15]; const float* node_ob = (const float*)d_in[16];
    const float* adj_ow  = (const float*)d_in[17]; const float* adj_ob  = (const float*)d_in[18];
    const float* tq_w = (const float*)d_in[19]; const float* tq_b = (const float*)d_in[20];
    const float* tk_w = (const float*)d_in[21]; const float* tk_b = (const float*)d_in[22];
    const float* tv_w = (const float*)d_in[23]; const float* tv_b = (const float*)d_in[24];
    const float* t_ow = (const float*)d_in[25]; const float* t_ob = (const float*)d_in[26];
    const float* t_f1w = (const float*)d_in[27]; const float* t_f1b = (const float*)d_in[28];
    const float* t_f2w = (const float*)d_in[29]; const float* t_f2b = (const float*)d_in[30];
    const float* t_n1s = (const float*)d_in[31]; const float* t_n1b = (const float*)d_in[32];
    const float* t_n2s = (const float*)d_in[33]; const float* t_n2b = (const float*)d_in[34];
    const float* f_nw1 = (const float*)d_in[35]; const float* f_nb1 = (const float*)d_in[36];
    const float* f_nw2 = (const float*)d_in[37]; const float* f_nb2 = (const float*)d_in[38];
    const float* f_aw1 = (const float*)d_in[39]; const float* f_ab1 = (const float*)d_in[40];
    const float* f_aw2 = (const float*)d_in[41]; const float* f_ab2 = (const float*)d_in[42];
    const float* n1s = (const float*)d_in[43]; const float* n1b = (const float*)d_in[44];
    const float* n2s = (const float*)d_in[45]; const float* n2b = (const float*)d_in[46];

    float* out_X   = (float*)d_out;                 // [B,N,D]   = 131072
    float* out_adj = (float*)d_out + B_ * N_ * D_;  // [B,N,N]   = 524288

    // ---- workspace carve-out ----
    float* ws = (float*)d_ws;
    size_t off = 0;
    auto alloc = [&](size_t n) { float* p = ws + off; off += n; return p; };
    const size_t BND = (size_t)B_ * N_ * D_;     // 131072
    float* aq1   = alloc(BND);  float* aq2 = alloc(BND);
    float* ah1   = alloc(BND);  float* ah2 = alloc(BND);
    float* feat  = alloc(BND * 5);               // [1024,640]
    float* Wt    = alloc((size_t)640 * 512);     // packed theta
    float* Hbuf  = alloc((size_t)1024 * 512);
    float* hproj = alloc(BND);
    float* X1    = alloc(BND);
    float* Qb    = alloc(BND); float* Kb = alloc(BND); float* Vb = alloc(BND);
    float* Amean = alloc(1024);
    float* modb  = alloc(4096);
    float* svec  = alloc(1024);
    float* Sbuf  = alloc((size_t)B_ * 4 * N_ * N_);  // 2M floats, reused temporally
    float* nodeb = alloc(BND);
    float* nodep = alloc(BND);
    float* X2    = alloc(BND);
    float* WtQ   = alloc((size_t)128 * 512);
    float* WtK   = alloc((size_t)128 * 512);
    float* WtV   = alloc((size_t)128 * 512);
    float* Qt    = alloc((size_t)1024 * 512);
    float* Kt    = alloc((size_t)1024 * 512);
    float* Vt    = alloc((size_t)1024 * 512);
    float* tout  = alloc((size_t)1024 * 512);
    float* tproj = alloc(BND);
    float* Xt    = alloc(BND);
    float* ffb   = alloc(BND);
    float* ffo   = alloc(BND);
    float* Xt2   = alloc(BND);
    float* fnb   = alloc(BND);

    // ---- weight repacks ----
    pack_hfo_kernel<<<(4 * 640 * 128 + 255) / 256, 256, 0, stream>>>(dgcn_theta, Wt, 4, 640, 128);
    pack_hfo_kernel<<<(4 * 128 * 128 + 255) / 256, 256, 0, stream>>>(tq_w, WtQ, 4, 128, 128);
    pack_hfo_kernel<<<(4 * 128 * 128 + 255) / 256, 256, 0, stream>>>(tk_w, WtK, 4, 128, 128);
    pack_hfo_kernel<<<(4 * 128 * 128 + 255) / 256, 256, 0, stream>>>(tv_w, WtV, 4, 128, 128);

    // ---- diffusion recursion: x1 = A@X ; x2 = 2*A@x1 - X ----
    for (int b = 0; b < B_; ++b) {
        const float* Aq = A_q + (size_t)b * N_ * N_;
        const float* Ah = A_h + (size_t)b * N_ * N_;
        const float* Xb = X + (size_t)b * N_ * D_;
        launch_gemm(stream, Aq, N_, Xb, D_, 0, aq1 + b * N_ * D_, D_, N_, D_, N_, 1.f, nullptr, nullptr, 0, 0.f, 0);
        launch_gemm(stream, Ah, N_, Xb, D_, 0, ah1 + b * N_ * D_, D_, N_, D_, N_, 1.f, nullptr, nullptr, 0, 0.f, 0);
        launch_gemm(stream, Aq, N_, aq1 + b * N_ * D_, D_, 0, aq2 + b * N_ * D_, D_, N_, D_, N_, 2.f, nullptr, Xb, D_, -1.f, 0);
        launch_gemm(stream, Ah, N_, ah1 + b * N_ * D_, D_, 0, ah2 + b * N_ * D_, D_, N_, D_, N_, 2.f, nullptr, Xb, D_, -1.f, 0);
    }
    interleave_feat_kernel<<<(B_ * N_ * D_ + 255) / 256, 256, 0, stream>>>(X, aq1, aq2, ah1, ah2, feat);

    // ---- DGCN heads + output proj + LN1 ----
    launch_gemm(stream, feat, 640, Wt, 512, 0, Hbuf, 512, 1024, 512, 640, 1.f, dgcn_bias, nullptr, 0, 0.f, 1);
    launch_gemm(stream, Hbuf, 512, dgcn_ow, 128, 0, hproj, 128, 1024, 128, 512, 1.f, dgcn_ob, nullptr, 0, 0.f, 0);
    ln_residual_kernel<<<1024, 128, 0, stream>>>(X, hproj, n1s, n1b, X1);

    // ---- spatial attention ----
    launch_gemm(stream, X1, 128, wq, 128, 0, Qb, 128, 1024, 128, 128, 1.f, bq, nullptr, 0, 0.f, 0);
    launch_gemm(stream, X1, 128, wk, 128, 0, Kb, 128, 1024, 128, 128, 1.f, bk, nullptr, 0, 0.f, 0);
    launch_gemm(stream, X1, 128, wv, 128, 0, Vb, 128, 1024, 128, 128, 1.f, bv, nullptr, 0, 0.f, 0);
    rowdot_kernel<<<1024, 128, 0, stream>>>(X1, adj_ow, svec);
    row_mean512_kernel<<<1024, 256, 0, stream>>>(A_q, Amean);
    mod_kernel<<<16, 256, 0, stream>>>(Amean, adjA_w, adjA_b, modb);

    const float sc_s = 0.17677669529663687f;   // 32^-0.5
    for (int b = 0; b < B_; ++b)
        for (int h = 0; h < H_; ++h) {
            float* S = Sbuf + (size_t)(b * H_ + h) * N_ * N_;
            launch_gemm(stream, Qb + b * N_ * D_ + h * 32, 128,
                        Kb + b * N_ * D_ + h * 32, 128, 1,
                        S, N_, N_, N_, 32, sc_s, nullptr, nullptr, 0, 0.f, 0);
        }
    softmax_row_kernel<<<B_ * H_ * N_, 256, 0, stream>>>(Sbuf, modb);
    for (int b = 0; b < B_; ++b)
        for (int h = 0; h < H_; ++h) {
            float* S = Sbuf + (size_t)(b * H_ + h) * N_ * N_;
            launch_gemm(stream, S, N_, Vb + b * N_ * D_ + h * 32, 128, 0,
                        nodeb + b * N_ * D_ + h * 32, 128, N_, 32, N_, 1.f, nullptr, nullptr, 0, 0.f, 0);
        }
    launch_gemm(stream, nodeb, 128, node_ow, 128, 0, nodep, 128, 1024, 128, 128, 1.f, node_ob, nullptr, 0, 0.f, 0);
    ln_residual_kernel<<<1024, 128, 0, stream>>>(X1, nodep, n2s, n2b, X2);

    // ---- temporal transformer ----
    launch_gemm(stream, X2, 128, WtQ, 512, 0, Qt, 512, 1024, 512, 128, 1.f, tq_b, nullptr, 0, 0.f, 0);
    launch_gemm(stream, X2, 128, WtK, 512, 0, Kt, 512, 1024, 512, 128, 1.f, tk_b, nullptr, 0, 0.f, 0);
    launch_gemm(stream, X2, 128, WtV, 512, 0, Vt, 512, 1024, 512, 128, 1.f, tv_b, nullptr, 0, 0.f, 0);
    const float sc_t = 0.08838834764831845f;   // 128^-0.5
    for (int b = 0; b < B_; ++b)
        for (int h = 0; h < HT_; ++h) {
            float* S = Sbuf + (size_t)(b * HT_ + h) * N_ * N_;
            launch_gemm(stream, Qt + (size_t)b * N_ * 512 + h * 128, 512,
                        Kt + (size_t)b * N_ * 512 + h * 128, 512, 1,
                        S, N_, N_, N_, 128, sc_t, nullptr, nullptr, 0, 0.f, 0);
        }
    softmax_row_kernel<<<B_ * HT_ * N_, 256, 0, stream>>>(Sbuf, nullptr);
    for (int b = 0; b < B_; ++b)
        for (int h = 0; h < HT_; ++h) {
            float* S = Sbuf + (size_t)(b * HT_ + h) * N_ * N_;
            launch_gemm(stream, S, N_, Vt + (size_t)b * N_ * 512 + h * 128, 512, 0,
                        tout + (size_t)b * N_ * 512 + h * 128, 512, N_, 128, N_, 1.f, nullptr, nullptr, 0, 0.f, 0);
        }
    launch_gemm(stream, tout, 512, t_ow, 128, 0, tproj, 128, 1024, 128, 512, 1.f, t_ob, nullptr, 0, 0.f, 0);
    ln_residual_kernel<<<1024, 128, 0, stream>>>(X2, tproj, t_n1s, t_n1b, Xt);
    launch_gemm(stream, Xt, 128, t_f1w, 128, 0, ffb, 128, 1024, 128, 128, 1.f, t_f1b, nullptr, 0, 0.f, 1);
    launch_gemm(stream, ffb, 128, t_f2w, 128, 0, ffo, 128, 1024, 128, 128, 1.f, t_f2b, nullptr, 0, 0.f, 0);
    ln_residual_kernel<<<1024, 128, 0, stream>>>(Xt, ffo, t_n2s, t_n2b, Xt2);

    // ---- block feed-forwards -> outputs ----
    launch_gemm(stream, Xt2, 128, f_nw1, 128, 0, fnb, 128, 1024, 128, 128, 1.f, f_nb1, nullptr, 0, 0.f, 1);
    launch_gemm(stream, fnb, 128, f_nw2, 128, 0, out_X, 128, 1024, 128, 128, 1.f, f_nb2, Xt2, 128, 1.f, 0);
    adj_out_kernel<<<(B_ * N_ * N_ + 255) / 256, 256, 0, stream>>>(A_q, svec, adj_ob,
                                                                   f_aw1, f_ab1, f_aw2, f_ab2,
                                                                   out_adj);
}
